// FlashSelfAttention_84198538871236
// MI455X (gfx1250) — matrix operand order repaired and checked by reading, not compiled, not run on hardware
//
#include <hip/hip_runtime.h>

// ---------------------------------------------------------------------------
// FlashSelfAttention for MI455X (gfx1250, wave32, WMMA bf16 16x16x32, TDM)
//   B=4, L=2048, D=1024, H=16, HD=64
// Pipeline: [cvt f32->bf16] -> [QKV GEMM] -> [flash attention] -> [proj GEMM]
// ---------------------------------------------------------------------------

typedef __bf16 bf16_t;
typedef __attribute__((ext_vector_type(16))) __bf16 v16bf;
typedef __attribute__((ext_vector_type(8)))  __bf16 v8bf;
typedef __attribute__((ext_vector_type(8)))  float  v8f;
typedef __attribute__((ext_vector_type(4)))  unsigned int v4u;
typedef __attribute__((ext_vector_type(8)))  int v8i;
typedef __attribute__((ext_vector_type(4)))  int v4i;

static constexpr int Bb = 4;
static constexpr int Lx = 2048;
static constexpr int Dd = 1024;
static constexpr int Hh = 16;
static constexpr int HDd = 64;

#define WMMA_BF16(a, b, c) \
  __builtin_amdgcn_wmma_f32_16x16x32_bf16(false, (a), false, (b), (short)0, (c), false, false)

// native hardware conversion (lowers to v_cvt_pk_bf16_f32 on gfx1250)
__device__ __forceinline__ bf16_t f2bf(float x) { return (bf16_t)x; }

// low 32 bits of a generic pointer to a __shared__ object = LDS byte address
__device__ __forceinline__ unsigned lds_addr_of(const void* p) {
  return (unsigned)(reinterpret_cast<unsigned long long>(p));
}

// ---------------------------------------------------------------------------
// TDM: issue a 2D bf16 tile load Global -> LDS (CDNA5 ISA).
//   tile_w elements per row (contiguous), tile_h rows, row stride in elements.
// Descriptor groups 2/3 (and the trailing group) are zero: 2-group descriptor,
// tensors up to 2D. Caller must be wave-uniform; completion via
// s_wait_tensorcnt + workgroup barrier.
// ---------------------------------------------------------------------------
__device__ __forceinline__ void tdm_load_tile_2d(unsigned lds_off, const void* gptr,
                                                 unsigned tile_w, unsigned tile_h,
                                                 unsigned row_stride) {
  unsigned long long ga = (unsigned long long)(reinterpret_cast<unsigned long long>(gptr));
  v4u g0;
  g0[0] = 1u;                                          // count=1, user descriptor
  g0[1] = lds_off;                                     // lds_addr (bytes)
  g0[2] = (unsigned)(ga & 0xFFFFFFFFu);                // global_addr[31:0]
  g0[3] = (unsigned)((ga >> 32) & 0x01FFFFFFu)         // global_addr[56:32]
        | (2u << 30);                                  // type=2 ("image")
  v8i g1;
  g1[0] = (int)(1u << 16);                             // wg_mask=0, data_size=1 (2B)
  g1[1] = (int)((row_stride & 0xFFFFu) << 16);         // tensor_dim0[15:0]
  g1[2] = (int)(((row_stride >> 16) & 0xFFFFu)         // tensor_dim0[31:16]
        | ((tile_h & 0xFFFFu) << 16));                 // tensor_dim1[15:0]
  g1[3] = (int)(((tile_h >> 16) & 0xFFFFu)             // tensor_dim1[31:16]
        | ((tile_w & 0xFFFFu) << 16));                 // tile_dim0
  g1[4] = (int)(tile_h & 0xFFFFu);                     // tile_dim1 (tile_dim2=0)
  g1[5] = (int)row_stride;                             // tensor_dim0_stride[31:0]
  g1[6] = 0;                                           // stride[47:32], dim1_stride lo
  g1[7] = 0;
  v4i z4 = {0, 0, 0, 0};
  v8i z8 = {0, 0, 0, 0, 0, 0, 0, 0};
  // 6-arg form (amdgpu-toolchain / clang-23 signature)
  __builtin_amdgcn_tensor_load_to_lds(g0, g1, z4, z4, z8, 0);
}

// ===========================================================================
// Kernel 0: bulk f32 -> bf16 conversion (8 elements / thread, 16B ld/st)
// ===========================================================================
__global__ __launch_bounds__(256) void cvt_bf16_kernel(
    const float* __restrict__ src, bf16_t* __restrict__ dst, int n) {
  int i = (blockIdx.x * blockDim.x + threadIdx.x) * 8;
  if (i >= n) return;
  float4 a = *(const float4*)(src + i);
  float4 b = *(const float4*)(src + i + 4);
  v8bf o;
  o[0] = (bf16_t)a.x; o[1] = (bf16_t)a.y; o[2] = (bf16_t)a.z; o[3] = (bf16_t)a.w;
  o[4] = (bf16_t)b.x; o[5] = (bf16_t)b.y; o[6] = (bf16_t)b.z; o[7] = (bf16_t)b.w;
  *(v8bf*)(dst + i) = o;
}

// --- A-operand fragment (16x32 bf16): lane<16 -> row lane, K {0..15};
//     lane>=16 -> row lane-16, K {16..31}. Source is row-major.
__device__ __forceinline__ v16bf load_a_bf16(const bf16_t* __restrict__ base,
                                             int ld, int m0, int k0) {
  int lane = threadIdx.x & 31;
  const bf16_t* p = base + (size_t)(m0 + (lane & 15)) * ld + (k0 + ((lane >> 4) << 4));
  v16bf f;
#pragma unroll
  for (int j = 0; j < 16; ++j) f[j] = p[j];
  return f;
}

// --- B-operand fragment (32x16 bf16): column n = n0 + (lane&15),
//     element j = B[k0 + (lane>=16)*16 + j][n]. For X*W^T, column n of B is
//     row n of W -> 16 consecutive elements of a row-major source.
__device__ __forceinline__ v16bf load_b_bf16(const bf16_t* __restrict__ base,
                                             int ld, int n0, int k0) {
  int lane = threadIdx.x & 31;
  const bf16_t* p = base + (size_t)(n0 + (lane & 15)) * ld + (k0 + ((lane >> 4) << 4));
  v16bf f;
#pragma unroll
  for (int j = 0; j < 16; ++j) f[j] = p[j];
  return f;
}

// --- B-operand fragment from an LDS tile (row-major [rows][ld] bf16)
__device__ __forceinline__ v16bf load_b_lds(const bf16_t* base, int ld, int n0, int k0) {
  int lane = threadIdx.x & 31;
  const bf16_t* p = base + (n0 + (lane & 15)) * ld + (k0 + ((lane >> 4) << 4));
  v16bf f;
#pragma unroll
  for (int j = 0; j < 16; ++j) f[j] = p[j];
  return f;
}

// --- A fragment reload from per-wave LDS tile (row-major 16x32 bf16)
__device__ __forceinline__ v16bf load_a_lds(const bf16_t* pb) {
  int lane = threadIdx.x & 31;
  const bf16_t* p = pb + (lane & 15) * 32 + ((lane >> 4) << 4);
  v16bf f;
#pragma unroll
  for (int j = 0; j < 16; ++j) f[j] = p[j];
  return f;
}

// ===========================================================================
// Kernel 1: qkv = x @ Wqkv^T ; scatter into Q[B,H,L,64], K[B,H,L,64],
//           Vt[B,H,64,L] (V transposed so it feeds WMMA B-operand directly).
// One wave computes a 16(M) x 64(N) tile, K-loop step 32, pure bf16 loads.
// ===========================================================================
__global__ __launch_bounds__(256) void qkv_proj_kernel(
    const bf16_t* __restrict__ xb, const bf16_t* __restrict__ wb,
    bf16_t* __restrict__ Q, bf16_t* __restrict__ K, bf16_t* __restrict__ Vt) {
  const int NT = (3 * Dd) / 64;  // 48 column groups
  int wave = blockIdx.x * 8 + (threadIdx.x >> 5);
  int mt = wave / NT, ng = wave % NT;
  int m0 = mt * 16, n0 = ng * 64;
  int lane = threadIdx.x & 31;

  v8f acc[4] = {};
  for (int k = 0; k < Dd; k += 32) {
    if (k + 32 < Dd) {
      __builtin_prefetch(xb + (size_t)(m0 + (lane & 15)) * Dd + k + 32, 0, 3);
      __builtin_prefetch(wb + (size_t)(n0 + (lane & 15)) * Dd + k + 32, 0, 3);
    }
    v16bf a  = load_a_bf16(xb, Dd, m0, k);
    v16bf b0 = load_b_bf16(wb, Dd, n0,      k);
    v16bf b1 = load_b_bf16(wb, Dd, n0 + 16, k);
    v16bf b2 = load_b_bf16(wb, Dd, n0 + 32, k);
    v16bf b3 = load_b_bf16(wb, Dd, n0 + 48, k);
    acc[0] = WMMA_BF16(a, b0, acc[0]);
    acc[1] = WMMA_BF16(a, b1, acc[1]);
    acc[2] = WMMA_BF16(a, b2, acc[2]);
    acc[3] = WMMA_BF16(a, b3, acc[3]);
  }

  int region = n0 / Dd;  // 0=Q 1=K 2=V (tile never straddles: 1024 % 64 == 0)
#pragma unroll
  for (int t = 0; t < 4; ++t) {
#pragma unroll
    for (int i = 0; i < 8; ++i) {
      int m = m0 + i + ((lane >> 4) << 3);         // C-layout row
      int n = n0 + t * 16 + (lane & 15);           // C-layout col
      int e = n - region * Dd;
      int h = e >> 6, d = e & 63;
      int b = m / Lx, pos = m - b * Lx;
      size_t bh = (size_t)(b * Hh + h);
      bf16_t val = f2bf(acc[t][i]);
      if (region == 0)      Q[(bh * Lx + pos) * HDd + d] = val;
      else if (region == 1) K[(bh * Lx + pos) * HDd + d] = val;
      else                  Vt[(bh * HDd + d) * Lx + pos] = val;
    }
  }
}

// ===========================================================================
// Kernel 2: flash attention. Block = 8 waves x 16 query rows = 128 rows of
// one (b,h). K/V tiles (32 keys) are DMA'd Global->LDS by the Tensor Data
// Mover, double-buffered, shared by all 8 waves. Per 32-key block:
// S = Q K^T (4 wmma), scale + padding mask, online softmax (shfl_xor row
// reductions in 16-lane groups), P staged through per-wave LDS for
// C-layout -> A-layout conversion, O += P V (4 wmma).
// ===========================================================================
__global__ __launch_bounds__(256) void flash_attn_kernel(
    const bf16_t* __restrict__ Q, const bf16_t* __restrict__ K,
    const bf16_t* __restrict__ Vt, const unsigned char* __restrict__ kpm,
    bf16_t* __restrict__ Y) {
  __shared__ bf16_t kbuf[2][32 * 64];   // 2 x 4 KB : 32 keys x 64 hd
  __shared__ bf16_t vbuf[2][64 * 32];   // 2 x 4 KB : 64 hd   x 32 keys
  __shared__ bf16_t pbuf[8][16 * 32];   // 8 KB     : per-wave P staging

  int wave = threadIdx.x >> 5, lane = threadIdx.x & 31;
  const int qblocks = Lx / 128;  // 16
  int bh = blockIdx.x / qblocks;
  int qb = blockIdx.x % qblocks;
  int b = bh / Hh, h = bh % Hh;
  int q0 = qb * 128 + wave * 16;

  const bf16_t* Qb = Q + (size_t)bh * Lx * HDd;
  const bf16_t* Kb = K + (size_t)bh * Lx * HDd;
  const bf16_t* Vb = Vt + (size_t)bh * HDd * Lx;
  const unsigned char* mrow = kpm + (size_t)b * Lx;

  v16bf qf0 = load_a_bf16(Qb, HDd, q0, 0);
  v16bf qf1 = load_a_bf16(Qb, HDd, q0, 32);

  // prologue: DMA first K/V tile into buffer 0 (one wave issues; TDM ignores EXEC)
  if (wave == 0) {
    tdm_load_tile_2d(lds_addr_of(&kbuf[0][0]), Kb, /*w*/ HDd, /*h*/ 32, /*stride*/ HDd);
    tdm_load_tile_2d(lds_addr_of(&vbuf[0][0]), Vb, /*w*/ 32, /*h*/ HDd, /*stride*/ Lx);
  }

  v8f o[4] = {};
  float m_i[8], l_i[8];
#pragma unroll
  for (int i = 0; i < 8; ++i) { m_i[i] = -__builtin_inff(); l_i[i] = 0.0f; }

  const float scale = 0.125f;  // 1/sqrt(64)

  for (int kb = 0; kb < Lx; kb += 32) {
    int cur = (kb >> 5) & 1;
    // release current tile to all waves: issuing wave drains TENSORcnt,
    // then the workgroup barrier publishes the LDS contents.
    if (wave == 0) __builtin_amdgcn_s_wait_tensorcnt(0);
    __syncthreads();
    // prefetch next K/V tile into the other buffer while we compute
    if (wave == 0 && kb + 32 < Lx) {
      tdm_load_tile_2d(lds_addr_of(&kbuf[cur ^ 1][0]),
                       Kb + (size_t)(kb + 32) * HDd, HDd, 32, HDd);
      tdm_load_tile_2d(lds_addr_of(&vbuf[cur ^ 1][0]),
                       Vb + (kb + 32), 32, HDd, Lx);
    }
    const bf16_t* kt = &kbuf[cur][0];  // [32 keys][64 hd]
    const bf16_t* vt = &vbuf[cur][0];  // [64 hd][32 keys]

    // ---- S tiles: 16q x 32k, K fragments from LDS
    v16bf kf00 = load_b_lds(kt, HDd, 0,  0);
    v16bf kf01 = load_b_lds(kt, HDd, 0,  32);
    v16bf kf10 = load_b_lds(kt, HDd, 16, 0);
    v16bf kf11 = load_b_lds(kt, HDd, 16, 32);
    v8f s0 = {}, s1 = {};
    s0 = WMMA_BF16(qf0, kf00, s0);
    s0 = WMMA_BF16(qf1, kf01, s0);
    s1 = WMMA_BF16(qf0, kf10, s1);
    s1 = WMMA_BF16(qf1, kf11, s1);

    float bias0 = mrow[kb + (lane & 15)]      ? -__builtin_inff() : 0.0f;
    float bias1 = mrow[kb + 16 + (lane & 15)] ? -__builtin_inff() : 0.0f;

    float p0[8], p1[8], alpha[8];
#pragma unroll
    for (int i = 0; i < 8; ++i) {
      float a0 = s0[i] * scale + bias0;
      float a1 = s1[i] * scale + bias1;
      // row-max across this 32-key block (rows live in 16-lane groups)
      float rm = fmaxf(a0, a1);
#pragma unroll
      for (int off = 8; off >= 1; off >>= 1)
        rm = fmaxf(rm, __shfl_xor(rm, off, 32));
      float mn = fmaxf(m_i[i], rm);
      bool dead = (mn == -__builtin_inff());  // whole row masked so far
      float al = dead ? 1.0f : __expf(m_i[i] - mn);
      float e0 = dead ? 0.0f : __expf(a0 - mn);
      float e1 = dead ? 0.0f : __expf(a1 - mn);
      float rs = e0 + e1;
#pragma unroll
      for (int off = 8; off >= 1; off >>= 1)
        rs += __shfl_xor(rs, off, 32);
      m_i[i] = mn;
      l_i[i] = l_i[i] * al + rs;
      alpha[i] = al;
      p0[i] = e0;
      p1[i] = e1;
    }

    // rescale running output accumulators
#pragma unroll
    for (int t = 0; t < 4; ++t)
#pragma unroll
      for (int i = 0; i < 8; ++i) o[t][i] *= alpha[i];

    // stage P (C-layout) into per-wave LDS tile, reload as A-operand fragment
    bf16_t* pb = pbuf[wave];
#pragma unroll
    for (int i = 0; i < 8; ++i) {
      int row = i + ((lane >> 4) << 3);
      int c = lane & 15;
      pb[row * 32 + c]      = f2bf(p0[i]);
      pb[row * 32 + 16 + c] = f2bf(p1[i]);
    }
    v16bf pf = load_a_lds(pb);  // same-wave DS ordering: no barrier needed

    // O += P * V  (V^T rows are head dims -> direct B-operand loads from LDS)
    v16bf vf0 = load_b_lds(vt, 32, 0,  0);
    v16bf vf1 = load_b_lds(vt, 32, 16, 0);
    v16bf vf2 = load_b_lds(vt, 32, 32, 0);
    v16bf vf3 = load_b_lds(vt, 32, 48, 0);
    o[0] = WMMA_BF16(pf, vf0, o[0]);
    o[1] = WMMA_BF16(pf, vf1, o[1]);
    o[2] = WMMA_BF16(pf, vf2, o[2]);
    o[3] = WMMA_BF16(pf, vf3, o[3]);
  }

  // epilogue: Y[b, q, h*64 + d] = O / l
#pragma unroll
  for (int t = 0; t < 4; ++t) {
#pragma unroll
    for (int i = 0; i < 8; ++i) {
      int row = q0 + i + ((lane >> 4) << 3);
      int col = h * HDd + t * 16 + (lane & 15);
      float val = o[t][i] / l_i[i];
      Y[((size_t)b * Lx + row) * Dd + col] = f2bf(val);
    }
  }
}

// ===========================================================================
// Kernel 3: out = Y @ Wproj^T  (f32 output)
// ===========================================================================
__global__ __launch_bounds__(256) void out_proj_kernel(
    const bf16_t* __restrict__ Y, const bf16_t* __restrict__ wb,
    float* __restrict__ out) {
  const int NT = Dd / 64;  // 16
  int wave = blockIdx.x * 8 + (threadIdx.x >> 5);
  int mt = wave / NT, ng = wave % NT;
  int m0 = mt * 16, n0 = ng * 64;
  int lane = threadIdx.x & 31;

  v8f acc[4] = {};
  for (int k = 0; k < Dd; k += 32) {
    if (k + 32 < Dd) {
      __builtin_prefetch(Y + (size_t)(m0 + (lane & 15)) * Dd + k + 32, 0, 3);
      __builtin_prefetch(wb + (size_t)(n0 + (lane & 15)) * Dd + k + 32, 0, 3);
    }
    v16bf a  = load_a_bf16(Y, Dd, m0, k);
    v16bf b0 = load_b_bf16(wb, Dd, n0,      k);
    v16bf b1 = load_b_bf16(wb, Dd, n0 + 16, k);
    v16bf b2 = load_b_bf16(wb, Dd, n0 + 32, k);
    v16bf b3 = load_b_bf16(wb, Dd, n0 + 48, k);
    acc[0] = WMMA_BF16(a, b0, acc[0]);
    acc[1] = WMMA_BF16(a, b1, acc[1]);
    acc[2] = WMMA_BF16(a, b2, acc[2]);
    acc[3] = WMMA_BF16(a, b3, acc[3]);
  }

#pragma unroll
  for (int t = 0; t < 4; ++t)
#pragma unroll
    for (int i = 0; i < 8; ++i) {
      int m = m0 + i + ((lane >> 4) << 3);
      int n = n0 + t * 16 + (lane & 15);
      out[(size_t)m * Dd + n] = acc[t][i];
    }
}

// ===========================================================================
extern "C" void kernel_launch(void* const* d_in, const int* in_sizes, int n_in,
                              void* d_out, int out_size, void* d_ws, size_t ws_size,
                              hipStream_t stream) {
  const float*         x     = (const float*)d_in[0];
  const unsigned char* kpm   = (const unsigned char*)d_in[1];  // bool mask, 1B/elt
  const float*         wqkv  = (const float*)d_in[2];          // [3D, D]
  const float*         wproj = (const float*)d_in[3];          // [D, D]
  float*               out   = (float*)d_out;

  char* ws = (char*)d_ws;
  const size_t seg = (size_t)Bb * Hh * Lx * HDd * sizeof(bf16_t);  // 16 MiB
  bf16_t* Q   = (bf16_t*)(ws);
  bf16_t* K   = (bf16_t*)(ws + seg);
  bf16_t* Vt  = (bf16_t*)(ws + 2 * seg);
  bf16_t* Y   = (bf16_t*)(ws + 3 * seg);
  bf16_t* xb  = (bf16_t*)(ws + 4 * seg);                            // 16 MiB
  bf16_t* wqb = (bf16_t*)(ws + 5 * seg);                            // 6 MiB
  bf16_t* wpb = (bf16_t*)(ws + 5 * seg + (size_t)3 * Dd * Dd * 2);  // 2 MiB

  const int n_x  = Bb * Lx * Dd;      // 8Mi
  const int n_wq = 3 * Dd * Dd;       // 3Mi
  const int n_wp = Dd * Dd;           // 1Mi

  // 0) one-shot f32 -> bf16 conversions (hoisted out of all GEMM loops)
  cvt_bf16_kernel<<<n_x  / (256 * 8), 256, 0, stream>>>(x,     xb,  n_x);
  cvt_bf16_kernel<<<n_wq / (256 * 8), 256, 0, stream>>>(wqkv,  wqb, n_wq);
  cvt_bf16_kernel<<<n_wp / (256 * 8), 256, 0, stream>>>(wproj, wpb, n_wp);

  // 1) QKV projection: (8192 x 1024) x (1024 x 3072)
  {
    int waves = (Bb * Lx / 16) * (3 * Dd / 64);  // 24576
    qkv_proj_kernel<<<waves / 8, 256, 0, stream>>>(xb, wqb, Q, K, Vt);
  }
  // 2) flash attention per (b, h, 128-row q block), K/V tiles via TDM
  {
    int blocks = Bb * Hh * (Lx / 128);           // 1024
    flash_attn_kernel<<<blocks, 256, 0, stream>>>(Q, K, Vt, kpm, Y);
  }
  // 3) output projection: (8192 x 1024) x (1024 x 1024)
  {
    int waves = (Bb * Lx / 16) * (Dd / 64);      // 8192
    out_proj_kernel<<<waves / 8, 256, 0, stream>>>(Y, wpb, out);
  }
}
